// CapsuleLayer_57629871178024
// MI455X (gfx1250) — matrix operand order repaired and checked by reading, not compiled, not run on hardware
//
#include <hip/hip_runtime.h>

typedef __bf16 bf16;
typedef __attribute__((ext_vector_type(16))) __bf16 v16bf;
typedef __attribute__((ext_vector_type(8)))  __bf16 v8bf;
typedef __attribute__((ext_vector_type(8)))  float  v8f;

constexpr int kB    = 32;    // batch
constexpr int kS    = 512;   // routes / sequence
constexpr int kC    = 16;    // capsules
constexpr int kDin  = 256;
constexpr int kDout = 64;
constexpr int kAL   = 264;   // padded LDS row stride (halves) for A tile  (16B aligned rows)
constexpr int kWL   = 264;   // padded LDS row stride (halves) for Wt tile

union Frag { v16bf v; v8bf h[2]; };

// ---------------------------------------------------------------------------
// Kernel 1: u_hat[b,c,s,o] = sum_i x[b,s,i] * W[c,s,i,o]
// One workgroup per (c,s): 32x64 GEMM, K=256, via bf16x3 split on
// v_wmma_f32_16x16x32_bf16 (hi*hi + hi*lo + lo*hi, f32 accumulate).
// ---------------------------------------------------------------------------
__global__ __launch_bounds__(256) void capsule_uhat_wmma(
    const float* __restrict__ x,      // [B, S, Din]
    const float* __restrict__ W,      // [C, S, Din, Dout]
    float* __restrict__ uhat) {       // [B, C, S, Dout]
  __shared__ bf16 Ah[kB * kAL];       // x tile hi   (32 x 256)
  __shared__ bf16 Al[kB * kAL];       // x tile lo
  __shared__ bf16 Bh[kDout * kWL];    // W tile hi, transposed: Wt[o][k] (64 x 256)
  __shared__ bf16 Bl[kDout * kWL];    // W tile lo

  const int bid = blockIdx.x;
  const int s   = bid & (kS - 1);     // s fastest: consecutive blocks stream W contiguously
  const int c   = bid >> 9;
  const int tid = threadIdx.x;

  // ---- stage A = x[:, s, :] as bf16 hi/lo --------------------------------
  for (int i = tid; i < kB * kDin; i += 256) {
    const int row = i >> 8;           // b
    const int k   = i & (kDin - 1);
    const float f = x[(size_t)row * kS * kDin + (size_t)s * kDin + k];
    const bf16 hi = (bf16)f;
    const bf16 lo = (bf16)(f - (float)hi);
    Ah[row * kAL + k] = hi;
    Al[row * kAL + k] = lo;
  }
  // ---- stage B = W[c, s, :, :] transposed -> Wt[o][k], bf16 hi/lo --------
  const float* wbase = W + ((size_t)c * kS + s) * kDin * kDout;
  for (int i = tid; i < kDin * kDout; i += 256) {     // coalesced read
    const int k = i >> 6;
    const int o = i & (kDout - 1);
    const float f = wbase[i];
    const bf16 hi = (bf16)f;
    const bf16 lo = (bf16)(f - (float)hi);
    Bh[o * kWL + k] = hi;
    Bl[o * kWL + k] = lo;
  }
  __syncthreads();

  // 8 waves: wave -> (mtile in 0..1, ntile in 0..3)
  const int wave = tid >> 5;
  const int lane = tid & 31;
  const int M0 = (wave & 1) * 16;
  const int N0 = (wave >> 1) * 16;

  // A 16x32 bf16 layout: lanes 0-15 M=0..15 K={0..7,16..23}; lanes 16-31 K={8..15,24..31}
  const int arow = M0 + (lane & 15);
  const int aoff = (lane >> 4) * 8;
  // B 32x16 bf16 layout: lane&15 = N; lanes 0-15 K={0..7,16..23}; lanes 16-31 K={8..15,24..31}
  const int bcol = N0 + (lane & 15);
  const int boff = (lane >> 4) * 8;

  v8f acc = {};
#pragma unroll
  for (int kb = 0; kb < kDin / 32; ++kb) {
    const int k0 = kb * 32;
    Frag ah, al, bh, bl;
    ah.h[0] = *(const v8bf*)&Ah[arow * kAL + k0 + aoff];
    ah.h[1] = *(const v8bf*)&Ah[arow * kAL + k0 + aoff + 16];
    al.h[0] = *(const v8bf*)&Al[arow * kAL + k0 + aoff];
    al.h[1] = *(const v8bf*)&Al[arow * kAL + k0 + aoff + 16];
    bh.h[0] = *(const v8bf*)&Bh[bcol * kWL + k0 + boff];
    bh.h[1] = *(const v8bf*)&Bh[bcol * kWL + k0 + boff + 16];
    bl.h[0] = *(const v8bf*)&Bl[bcol * kWL + k0 + boff];
    bl.h[1] = *(const v8bf*)&Bl[bcol * kWL + k0 + boff + 16];
    // bf16x3 emulation of f32 GEMM: hi*hi + hi*lo + lo*hi
    acc = __builtin_amdgcn_wmma_f32_16x16x32_bf16(false, ah.v, false, bh.v, (short)0, acc, false, false);
    acc = __builtin_amdgcn_wmma_f32_16x16x32_bf16(false, ah.v, false, bl.v, (short)0, acc, false, false);
    acc = __builtin_amdgcn_wmma_f32_16x16x32_bf16(false, al.v, false, bh.v, (short)0, acc, false, false);
  }

  // C/D layout: VGPR r, lane L -> M = r + 8*(L>=16), N = L&15
  const int n     = N0 + (lane & 15);
  const int mbase = M0 + ((lane >> 4) << 3);
#pragma unroll
  for (int r = 0; r < 8; ++r) {
    const int b = mbase + r;
    uhat[(((size_t)b * kC + c) * kS + s) * kDout + n] = acc[r];
  }
}

// ---------------------------------------------------------------------------
// Kernel 2: dynamic routing (3 iterations). One workgroup per batch element.
// b_ij / c_ij [16 x 512] live in LDS; each wave owns 2 capsules.
// ---------------------------------------------------------------------------
__global__ __launch_bounds__(256) void capsule_routing(
    const float* __restrict__ uhat,   // [B, C, S, Dout]
    float* __restrict__ out) {        // [B, C, Dout]
  __shared__ float bij[kC * kS];      // 32 KB
  __shared__ float cij[kC * kS];      // 32 KB
  __shared__ float vj[kC * kDout];    // 4 KB

  const int b    = blockIdx.x;
  const int tid  = threadIdx.x;
  const int wave = tid >> 5;
  const int lane = tid & 31;

  for (int i = tid; i < kC * kS; i += 256) bij[i] = 0.0f;
  __syncthreads();

  const float* ub = uhat + (size_t)b * kC * kS * kDout;

  for (int it = 0; it < 3; ++it) {
    // ---- softmax over capsules, per s ------------------------------------
    for (int s = tid; s < kS; s += 256) {
      float m = -1e30f;
#pragma unroll
      for (int cc = 0; cc < kC; ++cc) m = fmaxf(m, bij[cc * kS + s]);
      float e[kC];
      float sum = 0.0f;
#pragma unroll
      for (int cc = 0; cc < kC; ++cc) { e[cc] = __expf(bij[cc * kS + s] - m); sum += e[cc]; }
      const float inv = 1.0f / sum;
#pragma unroll
      for (int cc = 0; cc < kC; ++cc) cij[cc * kS + s] = e[cc] * inv;
    }
    __syncthreads();

    // ---- s_j = sum_s c_ij * u_hat, then squash ---------------------------
#pragma unroll
    for (int ci = 0; ci < 2; ++ci) {
      const int cc = wave * 2 + ci;
      const float* uc = ub + (size_t)cc * kS * kDout;
      float a0 = 0.0f, a1 = 0.0f;
      for (int s = 0; s < kS; ++s) {           // o = lane / lane+32 coalesced
        const float w = cij[cc * kS + s];
        a0 = fmaf(w, uc[s * kDout + lane], a0);
        a1 = fmaf(w, uc[s * kDout + lane + 32], a1);
      }
      // squash: ||s||^2 over 64 components via wave32 reduction
      float p = a0 * a0 + a1 * a1;
#pragma unroll
      for (int d = 16; d > 0; d >>= 1) p += __shfl_xor(p, d, 32);
      const float nrm   = sqrtf(p);
      const float scale = nrm / (1.0f + p);    // n/(1+n^2)
      const float v0 = a0 * scale, v1 = a1 * scale;
      vj[cc * kDout + lane]      = v0;
      vj[cc * kDout + lane + 32] = v1;
      if (it == 2) {
        out[((size_t)b * kC + cc) * kDout + lane]      = v0;
        out[((size_t)b * kC + cc) * kDout + lane + 32] = v1;
      }
    }
    __syncthreads();

    // ---- agreement: b_ij += u_hat . v ------------------------------------
    if (it < 2) {
#pragma unroll
      for (int ci = 0; ci < 2; ++ci) {
        const int cc = wave * 2 + ci;
        const float* uc = ub + (size_t)cc * kS * kDout;
        for (int s = lane; s < kS; s += 32) {  // one contiguous 256B row per lane
          const float* us = uc + (size_t)s * kDout;
          float dot = 0.0f;
#pragma unroll
          for (int o = 0; o < kDout; ++o) dot = fmaf(us[o], vj[cc * kDout + o], dot);
          bij[cc * kS + s] += dot;
        }
      }
      __syncthreads();
    }
  }
}

extern "C" void kernel_launch(void* const* d_in, const int* in_sizes, int n_in,
                              void* d_out, int out_size, void* d_ws, size_t ws_size,
                              hipStream_t stream) {
  const float* x = (const float*)d_in[0];   // [32, 512, 256]
  const float* W = (const float*)d_in[1];   // [16, 512, 256, 64]
  float* uhat = (float*)d_ws;               // 32*16*512*64 f32 = 64 MiB scratch
  float* out  = (float*)d_out;              // [32, 16, 64]

  capsule_uhat_wmma<<<kC * kS, 256, 0, stream>>>(x, W, uhat);
  capsule_routing<<<kB, 256, 0, stream>>>(uhat, out);
}
